// SimplicialMPLayer_2774548873281
// MI455X (gfx1250) — compile-verified, matrix operand order checked by reading, not run on hardware
//
#include <hip/hip_runtime.h>
#include <hip/hip_bf16.h>
#include <cstdint>
#include <cstddef>

typedef __attribute__((ext_vector_type(16))) _Float16 v16h;
typedef __attribute__((ext_vector_type(8)))  _Float16 h8;
typedef __attribute__((ext_vector_type(4)))  _Float16 h4;
typedef __attribute__((ext_vector_type(8)))  float    v8f;
typedef __attribute__((ext_vector_type(4)))  float    f4;

#define DIM 128
#define NPAD 136   // f16 LDS row stride (halves): mult. of 8 -> 16B-aligned h8 loads
#define APAD 132   // f32 LDS row stride (floats): mult. of 4 -> 16B-aligned f4 loads

// ---------------------------------------------------------------------------
// CDNA5 async copy global->LDS (ASYNCcnt path).  LDS operand is the low 32
// bits of the flat pointer.
// ---------------------------------------------------------------------------
__device__ __forceinline__ void async_copy_b128(void* lds, const void* gptr) {
    asm volatile("global_load_async_to_lds_b128 %0, %1, off"
                 :: "v"((uint32_t)(uintptr_t)lds), "v"(gptr)
                 : "memory");
}
__device__ __forceinline__ void wait_async0() {
    asm volatile("s_wait_asynccnt 0" ::: "memory");
}

// ---------------------------------------------------------------------------
// WMMA helpers.  CDNA5 16-bit fragment layout (CDNA5 ISA):
//   A (16x32, MxK): lane L -> m = L&15 ; element e -> k = (e>>3)*16 + ((L>>4)*8) + (e&7)
//   B (32x16, KxN): lane L -> n = L&15 ; same k mapping (B stored as Wt[n][k])
//   C/D (16x16 f32): lane L -> n = L&15 ; element r -> m = r + (L>=16 ? 8 : 0)
// ---------------------------------------------------------------------------
__device__ __forceinline__ v8f wmma_f16(v16h a, v16h b, v8f c) {
    return __builtin_amdgcn_wmma_f32_16x16x32_f16(false, a, false, b, (short)0, c,
                                                  false, false);
}

__device__ __forceinline__ v16h load_frag_f16(const _Float16* base, int ld, int k0, int lane) {
    int rc = lane & 15;
    int kb = (lane >> 4) << 3;
    const _Float16* p = base + (size_t)rc * ld + k0 + kb;
    h8 lo = *(const h8*)p;
    h8 hi = *(const h8*)(p + 16);
    v16h r;
#pragma unroll
    for (int i = 0; i < 8; ++i) { r[i] = lo[i]; r[8 + i] = hi[i]; }
    return r;
}

__device__ __forceinline__ v16h load_frag_f32cvt(const float* base, int ld, int k0, int lane) {
    int rc = lane & 15;
    int kb = (lane >> 4) << 3;
    const float* p = base + (size_t)rc * ld + k0 + kb;
    f4 a0 = *(const f4*)p;
    f4 a1 = *(const f4*)(p + 4);
    f4 a2 = *(const f4*)(p + 16);
    f4 a3 = *(const f4*)(p + 20);
    v16h r;
#pragma unroll
    for (int i = 0; i < 4; ++i) {
        r[i]      = (_Float16)a0[i];
        r[4 + i]  = (_Float16)a1[i];
        r[8 + i]  = (_Float16)a2[i];
        r[12 + i] = (_Float16)a3[i];
    }
    return r;
}

__device__ __forceinline__ float silu_f(float v) {
    return v * (1.0f / (1.0f + __expf(-v)));
}

// ---------------------------------------------------------------------------
// Weight prep: f32 row-major (K x N) -> f16 transposed Wt[n*K + k]
// ---------------------------------------------------------------------------
__global__ __launch_bounds__(256) void wtrans_kernel(const float* __restrict__ src,
                                                     _Float16* __restrict__ dst,
                                                     int K, int N) {
    int idx = blockIdx.x * 256 + threadIdx.x;
    if (idx >= K * N) return;
    int k = idx / N, n = idx - k * N;
    dst[(size_t)n * K + k] = (_Float16)src[idx];
}

// ---------------------------------------------------------------------------
// Edge scatter-add:  agg[dst[e]] += x_src[src[e]]   (one wave per edge)
// ---------------------------------------------------------------------------
__global__ __launch_bounds__(256) void scatter_add_kernel(const float* __restrict__ xsrc,
                                                          const int* __restrict__ src,
                                                          const int* __restrict__ dst,
                                                          float* __restrict__ agg, int E) {
    int gid  = blockIdx.x * 256 + threadIdx.x;
    int e    = gid >> 5;
    int lane = gid & 31;
    if (e >= E) return;
    int s = src[e];
    int d = dst[e];
    f4 v = *((const f4*)(xsrc + (size_t)s * DIM) + lane);
    float* pd = agg + (size_t)d * DIM + lane * 4;
#pragma unroll
    for (int i = 0; i < 4; ++i)
        __hip_atomic_fetch_add(pd + i, v[i], __ATOMIC_RELAXED, __HIP_MEMORY_SCOPE_AGENT);
}

// ---------------------------------------------------------------------------
// Fused GIN MLP:  mes = silu(agg @ W1 + b1) @ W2 + b2   (f16 out)
// block = 4 waves, each wave owns 16 rows x 128 cols
// ---------------------------------------------------------------------------
__global__ __launch_bounds__(128) void gin_mlp_kernel(const float* __restrict__ agg,
                                                      const _Float16* __restrict__ W1t,
                                                      const float* __restrict__ b1,
                                                      const _Float16* __restrict__ W2t,
                                                      const float* __restrict__ b2,
                                                      _Float16* __restrict__ mes, int nrows) {
    __shared__ __align__(16) float    abuf[4][16][APAD];
    __shared__ __align__(16) _Float16 hbuf[4][16][NPAD];
    int lane = threadIdx.x & 31;
    int wave = threadIdx.x >> 5;
    int m0 = blockIdx.x * 64 + wave * 16;
    if (m0 + 16 > nrows) m0 = nrows - 16;          // defensive clamp (N % 64 == 0 here)

    // --- async-stage this wave's 16x128 f32 A-tile into LDS (coalesced rows) ---
    const float* arow = agg + (size_t)m0 * DIM;
#pragma unroll
    for (int r = 0; r < 16; ++r)
        async_copy_b128(&abuf[wave][r][lane * 4], arow + r * DIM + lane * 4);
    wait_async0();

    v8f acc[8];
#pragma unroll
    for (int j = 0; j < 8; ++j)
#pragma unroll
        for (int r = 0; r < 8; ++r) acc[j][r] = 0.0f;

    const float* atile = &abuf[wave][0][0];
#pragma unroll
    for (int kk = 0; kk < 4; ++kk) {
        v16h a = load_frag_f32cvt(atile, APAD, kk * 32, lane);
#pragma unroll
        for (int j = 0; j < 8; ++j) {
            v16h b = load_frag_f16(W1t + (size_t)j * 16 * DIM, DIM, kk * 32, lane);
            acc[j] = wmma_f16(a, b, acc[j]);
        }
    }

    int n_lane = lane & 15;
    int mtop   = (lane >> 4) << 3;
#pragma unroll
    for (int j = 0; j < 8; ++j) {
        float bias = b1[j * 16 + n_lane];
#pragma unroll
        for (int r = 0; r < 8; ++r)
            hbuf[wave][mtop + r][j * 16 + n_lane] = (_Float16)silu_f(acc[j][r] + bias);
    }
    __syncthreads();

    v8f acc2[8];
#pragma unroll
    for (int j = 0; j < 8; ++j)
#pragma unroll
        for (int r = 0; r < 8; ++r) acc2[j][r] = 0.0f;

    const _Float16* hrow = &hbuf[wave][0][0];
#pragma unroll
    for (int kk = 0; kk < 4; ++kk) {
        v16h a = load_frag_f16(hrow, NPAD, kk * 32, lane);
#pragma unroll
        for (int j = 0; j < 8; ++j) {
            v16h b = load_frag_f16(W2t + (size_t)j * 16 * DIM, DIM, kk * 32, lane);
            acc2[j] = wmma_f16(a, b, acc2[j]);
        }
    }

    // --- bounce result through LDS (wave-local) for coalesced b64 stores ---
#pragma unroll
    for (int j = 0; j < 8; ++j) {
        float bias = b2[j * 16 + n_lane];
#pragma unroll
        for (int r = 0; r < 8; ++r)
            hbuf[wave][mtop + r][j * 16 + n_lane] = (_Float16)(acc2[j][r] + bias);
    }
#pragma unroll
    for (int r = 0; r < 16; ++r) {
        h4 v = *(const h4*)&hbuf[wave][r][lane * 4];
        *(h4*)(mes + (size_t)(m0 + r) * DIM + lane * 4) = v;
    }
}

// ---------------------------------------------------------------------------
// Fused update MLP:  out = x + silu([x|mesA|mesB] @ W1 + b1) @ W2 + b2  (f32 out)
// ---------------------------------------------------------------------------
__global__ __launch_bounds__(128) void update_mlp_kernel(const float* __restrict__ x,
                                                         const _Float16* __restrict__ mesA,
                                                         const _Float16* __restrict__ mesB,
                                                         const _Float16* __restrict__ W1t,
                                                         const float* __restrict__ b1,
                                                         const _Float16* __restrict__ W2t,
                                                         const float* __restrict__ b2,
                                                         float* __restrict__ out,
                                                         int nrows, int factor) {
    __shared__ __align__(16) float    abuf[4][16][APAD];   // x tile, later f32 out tile
    __shared__ __align__(16) _Float16 hbuf[4][16][NPAD];
    int lane = threadIdx.x & 31;
    int wave = threadIdx.x >> 5;
    int m0 = blockIdx.x * 64 + wave * 16;
    if (m0 + 16 > nrows) m0 = nrows - 16;

    int KT = factor * DIM;

    // --- async-stage this wave's 16x128 f32 x-tile into LDS ---
    const float* xrow = x + (size_t)m0 * DIM;
#pragma unroll
    for (int r = 0; r < 16; ++r)
        async_copy_b128(&abuf[wave][r][lane * 4], xrow + r * DIM + lane * 4);
    wait_async0();

    v8f acc[8];
#pragma unroll
    for (int j = 0; j < 8; ++j)
#pragma unroll
        for (int r = 0; r < 8; ++r) acc[j][r] = 0.0f;

    const float* atile = &abuf[wave][0][0];
    for (int kk = 0; kk < factor * 4; ++kk) {
        int seg = kk >> 2;
        int kl  = (kk & 3) * 32;
        v16h a;
        if (seg == 0) {
            a = load_frag_f32cvt(atile, APAD, kl, lane);
        } else {
            const _Float16* msrc = (seg == 1) ? mesA : mesB;
            a = load_frag_f16(msrc + (size_t)m0 * DIM, DIM, kl, lane);
        }
        int kglob = seg * DIM + kl;
#pragma unroll
        for (int j = 0; j < 8; ++j) {
            v16h b = load_frag_f16(W1t + (size_t)j * 16 * KT, KT, kglob, lane);
            acc[j] = wmma_f16(a, b, acc[j]);
        }
    }

    int n_lane = lane & 15;
    int mtop   = (lane >> 4) << 3;
#pragma unroll
    for (int j = 0; j < 8; ++j) {
        float bias = b1[j * 16 + n_lane];
#pragma unroll
        for (int r = 0; r < 8; ++r)
            hbuf[wave][mtop + r][j * 16 + n_lane] = (_Float16)silu_f(acc[j][r] + bias);
    }
    __syncthreads();

    v8f acc2[8];
#pragma unroll
    for (int j = 0; j < 8; ++j)
#pragma unroll
        for (int r = 0; r < 8; ++r) acc2[j][r] = 0.0f;

    const _Float16* hrow = &hbuf[wave][0][0];
#pragma unroll
    for (int kk = 0; kk < 4; ++kk) {
        v16h a = load_frag_f16(hrow, NPAD, kk * 32, lane);
#pragma unroll
        for (int j = 0; j < 8; ++j) {
            v16h b = load_frag_f16(W2t + (size_t)j * 16 * DIM, DIM, kk * 32, lane);
            acc2[j] = wmma_f16(a, b, acc2[j]);
        }
    }

    // --- bounce f32 result through LDS (reuse abuf), coalesced residual+store ---
#pragma unroll
    for (int j = 0; j < 8; ++j) {
        float bias = b2[j * 16 + n_lane];
#pragma unroll
        for (int r = 0; r < 8; ++r)
            abuf[wave][mtop + r][j * 16 + n_lane] = acc2[j][r] + bias;
    }
#pragma unroll
    for (int r = 0; r < 16; ++r) {
        f4 v  = *(const f4*)&abuf[wave][r][lane * 4];
        f4 xr = *(const f4*)(x + (size_t)(m0 + r) * DIM + lane * 4);
#pragma unroll
        for (int i = 0; i < 4; ++i) v[i] += xr[i];
        *(f4*)(out + (size_t)(m0 + r) * DIM + lane * 4) = v;
    }
}

// ---------------------------------------------------------------------------
// Host launcher
// ---------------------------------------------------------------------------
extern "C" void kernel_launch(void* const* d_in, const int* in_sizes, int n_in,
                              void* d_out, int out_size, void* d_ws, size_t ws_size,
                              hipStream_t stream) {
    const int N = in_sizes[0] / DIM;
    const int E = in_sizes[3];

    const float* x[3] = { (const float*)d_in[0], (const float*)d_in[1], (const float*)d_in[2] };
    const int* esrc[5] = { (const int*)d_in[3], (const int*)d_in[5], (const int*)d_in[7],
                           (const int*)d_in[9], (const int*)d_in[11] };
    const int* edst[5] = { (const int*)d_in[4], (const int*)d_in[6], (const int*)d_in[8],
                           (const int*)d_in[10], (const int*)d_in[12] };
    const float* mpW1 = (const float*)d_in[13];
    const float* mpb1 = (const float*)d_in[14];
    const float* mpW2 = (const float*)d_in[15];
    const float* mpb2 = (const float*)d_in[16];
    const float* upW1[3] = { (const float*)d_in[17], (const float*)d_in[21], (const float*)d_in[25] };
    const float* upb1[3] = { (const float*)d_in[18], (const float*)d_in[22], (const float*)d_in[26] };
    const float* upW2[3] = { (const float*)d_in[19], (const float*)d_in[23], (const float*)d_in[27] };
    const float* upb2[3] = { (const float*)d_in[20], (const float*)d_in[24], (const float*)d_in[28] };

    const int srcDim[5] = { 0, 0, 1, 1, 2 };   // ADJ = 00, 01, 11, 12, 22
    const int dstDim[5] = { 0, 1, 1, 2, 2 };
    const int upFactor[3] = { 2, 3, 3 };

    const size_t ND = (size_t)N * DIM;
    const size_t WW = (size_t)DIM * DIM;       // 16384

    // --- workspace carve-up ---
    float*     agg  = (float*)d_ws;                       // 5 * ND  f32
    _Float16*  mes  = (_Float16*)(agg + 5 * ND);          // 5 * ND  f16
    _Float16*  wp   = mes + 5 * ND;                       // f16 weights
    _Float16*  mpW1t = wp;                 wp += 5 * WW;
    _Float16*  mpW2t = wp;                 wp += 5 * WW;
    _Float16*  upW1t[3]; _Float16* upW2t[3];
    for (int d = 0; d < 3; ++d) { upW1t[d] = wp; wp += (size_t)upFactor[d] * WW; }
    for (int d = 0; d < 3; ++d) { upW2t[d] = wp; wp += WW; }

    // --- 1. weight transpose/convert ---
    {
        int blocks = (int)((WW + 255) / 256);
        for (int t = 0; t < 5; ++t) {
            wtrans_kernel<<<blocks, 256, 0, stream>>>(mpW1 + t * WW, mpW1t + t * WW, DIM, DIM);
            wtrans_kernel<<<blocks, 256, 0, stream>>>(mpW2 + t * WW, mpW2t + t * WW, DIM, DIM);
        }
        for (int d = 0; d < 3; ++d) {
            int K1 = upFactor[d] * DIM;
            wtrans_kernel<<<(int)(((size_t)K1 * DIM + 255) / 256), 256, 0, stream>>>(
                upW1[d], upW1t[d], K1, DIM);
            wtrans_kernel<<<blocks, 256, 0, stream>>>(upW2[d], upW2t[d], DIM, DIM);
        }
    }

    // --- 2. agg init (GIN eps=0 self term) + edge scatter ---
    for (int t = 0; t < 5; ++t)
        hipMemcpyAsync(agg + t * ND, x[dstDim[t]], ND * sizeof(float),
                       hipMemcpyDeviceToDevice, stream);
    {
        int blocks = (E * 32 + 255) / 256;
        for (int t = 0; t < 5; ++t)
            scatter_add_kernel<<<blocks, 256, 0, stream>>>(x[srcDim[t]], esrc[t], edst[t],
                                                           agg + t * ND, E);
    }

    // --- 3. fused GIN MLPs ---
    {
        int blocks = (N + 63) / 64;
        for (int t = 0; t < 5; ++t)
            gin_mlp_kernel<<<blocks, 128, 0, stream>>>(agg + t * ND,
                                                       mpW1t + t * WW, mpb1 + t * DIM,
                                                       mpW2t + t * WW, mpb2 + t * DIM,
                                                       mes + t * ND, N);
    }

    // --- 4. fused update MLPs (+residual) ---
    {
        int blocks = (N + 63) / 64;
        float* out = (float*)d_out;
        // dim 0: [x0 | mes00]
        update_mlp_kernel<<<blocks, 128, 0, stream>>>(x[0], mes + 0 * ND, mes + 0 * ND,
                                                      upW1t[0], upb1[0], upW2t[0], upb2[0],
                                                      out + 0 * ND, N, 2);
        // dim 1: [x1 | mes01 | mes11]
        update_mlp_kernel<<<blocks, 128, 0, stream>>>(x[1], mes + 1 * ND, mes + 2 * ND,
                                                      upW1t[1], upb1[1], upW2t[1], upb2[1],
                                                      out + 1 * ND, N, 3);
        // dim 2: [x2 | mes12 | mes22]
        update_mlp_kernel<<<blocks, 128, 0, stream>>>(x[2], mes + 3 * ND, mes + 4 * ND,
                                                      upW1t[2], upb1[2], upW2t[2], upb2[2],
                                                      out + 2 * ND, N, 3);
    }
}
